// Attention_29781303230706
// MI455X (gfx1250) — matrix operand order repaired and checked by reading, not compiled, not run on hardware
//
#include <hip/hip_runtime.h>

#define BATCH 4
#define S_LEN 2048
#define D_DIM 1024

typedef __bf16 bf16_t;
typedef __attribute__((ext_vector_type(16))) __bf16 v16bf;
typedef __attribute__((ext_vector_type(8)))  __bf16 v8bf;
typedef __attribute__((ext_vector_type(8)))  float  v8f;
typedef __attribute__((ext_vector_type(4)))  float  v4f;
typedef __attribute__((ext_vector_type(4)))  unsigned int v4u;
typedef __attribute__((ext_vector_type(8)))  int v8i;
typedef __attribute__((ext_vector_type(4)))  int v4i;

__device__ inline bf16_t f2bf(float f) { return (bf16_t)f; }

__device__ inline v8f wmma_bf16(v16bf a, v16bf b, v8f c) {
  return __builtin_amdgcn_wmma_f32_16x16x32_bf16(false, a, false, b, (short)0, c,
                                                 false, false);
}

// ---- Tensor Data Mover: 2-D bf16 tile (rows x cols) global -> LDS ----
// D# per CDNA5 ISA ch.8: group0 = {count/lds_addr/global_addr/type=2},
// group1 = {data_size=2B, tensor dims, tile dims, dim0 stride}, groups 2/3 = 0.
__device__ inline void tdm_load_2d_bf16(unsigned lds_byte_off, const void* gptr,
                                        unsigned rows, unsigned cols,
                                        unsigned stride_elems) {
  unsigned long long ga = (unsigned long long)(uintptr_t)gptr;
  v4u g0;
  g0[0] = 1u;                                        // count=1, user descriptor
  g0[1] = lds_byte_off;                              // lds_addr (bytes)
  g0[2] = (unsigned)(ga & 0xFFFFFFFFu);              // global_addr[31:0]
  g0[3] = (unsigned)((ga >> 32) & 0x01FFFFFFu)       // global_addr[56:32]
          | 0x80000000u;                             // type=2 ("image")
  v8i g1;
  g1[0] = 0x00010000;                                // data_size=1 -> 2 bytes
  g1[1] = (int)((cols & 0xFFFFu) << 16);             // tensor_dim0[15:0]
  g1[2] = (int)(((cols >> 16) & 0xFFFFu) |           // tensor_dim0[31:16]
                ((rows & 0xFFFFu) << 16));           // tensor_dim1[15:0]
  g1[3] = (int)(((rows >> 16) & 0xFFFFu) |           // tensor_dim1[31:16]
                ((cols & 0xFFFFu) << 16));           // tile_dim0
  g1[4] = (int)(rows & 0xFFFFu);                     // tile_dim1 (tile_dim2=0)
  g1[5] = (int)stride_elems;                         // tensor_dim0_stride lo32
  g1[6] = 0;                                         // stride0 hi / stride1 lo
  g1[7] = 0;
  v4i z = {0, 0, 0, 0};
#if __has_include(<hip/amd_detail/amd_gfx1250_TDM.h>)
  v8i z8 = {0, 0, 0, 0, 0, 0, 0, 0};
  __builtin_amdgcn_tensor_load_to_lds(g0, g1, z, z, z8, 0);
#else
  __builtin_amdgcn_tensor_load_to_lds(g0, g1, z, z, 0);
#endif
}

// A-matrix 16x32 bf16 fragment, row-major source (row stride ld).
__device__ inline v16bf load_a_bf16(const bf16_t* base, int ld) {
  int lane = threadIdx.x & 31;
  int m = lane & 15;
  int k0 = (lane >> 4) ? 8 : 0;
  const bf16_t* p = base + m * ld + k0;
  v8bf lo = *reinterpret_cast<const v8bf*>(p);
  v8bf hi = *reinterpret_cast<const v8bf*>(p + 16);
  v16bf a;
#pragma unroll
  for (int i = 0; i < 8; ++i) { a[i] = lo[i]; a[8 + i] = hi[i]; }
  return a;
}

// B-matrix 32x16 fragment where B[kk][n] = mem[n*ld + kk] (K contiguous/lane).
__device__ inline v16bf load_bT_bf16(const bf16_t* base, int ld) {
  int lane = threadIdx.x & 31;
  const bf16_t* p = base + (size_t)(lane & 15) * ld + ((lane >> 4) ? 8 : 0);
  v8bf lo = *reinterpret_cast<const v8bf*>(p);
  v8bf hi = *reinterpret_cast<const v8bf*>(p + 16);
  v16bf b;
#pragma unroll
  for (int i = 0; i < 8; ++i) { b[i] = lo[i]; b[8 + i] = hi[i]; }
  return b;
}

// ---------------- Kernel 0a: x fp32 -> bf16 (coalesced) ----------------
__global__ __launch_bounds__(256) void xconv_kernel(const float* __restrict__ x,
                                                    bf16_t* __restrict__ xb) {
  size_t i = ((size_t)blockIdx.x * 256 + threadIdx.x) * 8;
  v4f f0 = *reinterpret_cast<const v4f*>(x + i);
  v4f f1 = *reinterpret_cast<const v4f*>(x + i + 4);
  v8bf o;
#pragma unroll
  for (int j = 0; j < 4; ++j) { o[j] = f2bf(f0[j]); o[4 + j] = f2bf(f1[j]); }
  *reinterpret_cast<v8bf*>(xb + i) = o;
}

// ------- Kernel 0b: transpose+convert Wq/Wk/Wv (fp32 [D][D] -> bf16 W^T) -----
__global__ __launch_bounds__(256) void wconv_kernel(
    const float* __restrict__ Wq, const float* __restrict__ Wk,
    const float* __restrict__ Wv, bf16_t* __restrict__ Wqt,
    bf16_t* __restrict__ Wkt, bf16_t* __restrict__ Wvt) {
  __shared__ bf16_t tile[64][65];
  const float* src = (blockIdx.y == 0) ? Wq : (blockIdx.y == 1) ? Wk : Wv;
  bf16_t* dst = (blockIdx.y == 0) ? Wqt : (blockIdx.y == 1) ? Wkt : Wvt;
  const int c0 = (blockIdx.x & 15) * 64;
  const int r0 = (blockIdx.x >> 4) * 64;
  const int f = threadIdx.x & 63;
  const int s4 = threadIdx.x >> 6;
#pragma unroll
  for (int i = 0; i < 16; ++i) {
    int r = s4 + i * 4;
    tile[f][r] = f2bf(src[(size_t)(r0 + r) * D_DIM + c0 + f]);
  }
  __syncthreads();
#pragma unroll
  for (int i = 0; i < 16; ++i) {
    int cc = s4 + i * 4;
    dst[(size_t)(c0 + cc) * D_DIM + r0 + f] = tile[cc][f];
  }
}

// ---------------- Kernel 1: fused QKV projection ----------------
// Block owns one 16-row M-tile; its 16x1024 x tile is TDM'd into LDS once.
// Wave w computes cols [w*128, w*128+128) for Q, K and V (24 accumulators).
// V is stored TRANSPOSED [B][D][S] with contiguous v8bf stores.
__global__ __launch_bounds__(256) void qkv_kernel(
    const bf16_t* __restrict__ xb,
    const bf16_t* __restrict__ Wqt, const bf16_t* __restrict__ Wkt,
    const bf16_t* __restrict__ Wvt,
    const float* __restrict__ bq, const float* __restrict__ bk,
    const float* __restrict__ bv,
    bf16_t* __restrict__ Qb, bf16_t* __restrict__ Kb, bf16_t* __restrict__ Vt) {
  __shared__ bf16_t xtile[16 * D_DIM];          // 32 KB, LDS offset 0
  const int wave = threadIdx.x >> 5;
  const int lane = threadIdx.x & 31;
  const int row0 = blockIdx.x * 16;             // 512 blocks over B*S
  const int col0 = wave * 128;

  if (threadIdx.x < 32) {                       // wave 0 issues the TDM
    tdm_load_2d_bf16(0u, xb + (size_t)row0 * D_DIM, 16u, (unsigned)D_DIM,
                     (unsigned)D_DIM);
    __builtin_amdgcn_s_wait_tensorcnt(0);
  }
  __syncthreads();

  v8f accQ[8] = {}; v8f accK[8] = {}; v8f accV[8] = {};
  for (int kb = 0; kb < D_DIM; kb += 32) {
    v16bf a = load_a_bf16(xtile + kb, D_DIM);   // ds_load_b128 x2
#pragma unroll
    for (int sub = 0; sub < 8; ++sub) {
      const size_t wb = (size_t)(col0 + sub * 16) * D_DIM + kb;  // W^T[n][kk]
      accQ[sub] = wmma_bf16(a, load_bT_bf16(Wqt + wb, D_DIM), accQ[sub]);
      accK[sub] = wmma_bf16(a, load_bT_bf16(Wkt + wb, D_DIM), accK[sub]);
      accV[sub] = wmma_bf16(a, load_bT_bf16(Wvt + wb, D_DIM), accV[sub]);
    }
  }
  const int mOff = (lane >> 4) ? 8 : 0;
  const int bb = row0 >> 11;                    // batch
  const int srow = row0 & (S_LEN - 1);
#pragma unroll
  for (int sub = 0; sub < 8; ++sub) {
    const int n = col0 + sub * 16 + (lane & 15);
    const float biq = bq[n], bik = bk[n], biv = bv[n];
#pragma unroll
    for (int r = 0; r < 8; ++r) {
      const size_t idx = (size_t)(row0 + r + mOff) * D_DIM + n;
      Qb[idx] = f2bf(accQ[sub][r] + biq);
      Kb[idx] = f2bf(accK[sub][r] + bik);
    }
    v8bf vv;
#pragma unroll
    for (int r = 0; r < 8; ++r) vv[r] = f2bf(accV[sub][r] + biv);
    *reinterpret_cast<v8bf*>(Vt + ((size_t)bb * D_DIM + n) * S_LEN + srow + mOff) = vv;
  }
}

// ---------------- Kernel 2: fused causal attention ----------------
// Block owns one 16-query tile; the 16x1024 Q tile is TDM'd into LDS once
// (frees ~256 VGPRs/wave vs register caching). Per 128-key block: wave w
// computes its 16-key score subtile (WMMA), exp + causal mask, row sums into
// LDS, stages P bf16 in LDS; then each wave accumulates its 16x128 d-slab of
// P@V from V^T. Normalize by row sums at the end.
__global__ __launch_bounds__(256) void attn_kernel(
    const bf16_t* __restrict__ Qb, const bf16_t* __restrict__ Kb,
    const bf16_t* __restrict__ Vt, float* __restrict__ out) {
  __shared__ bf16_t Qlds[16 * D_DIM];           // 32 KB, LDS offset 0
  __shared__ bf16_t Plds[16 * 128];             // 4 KB
  __shared__ float lsum[16];
  const int wave = threadIdx.x >> 5;
  const int lane = threadIdx.x & 31;
  const int b  = blockIdx.x >> 7;
  const int q0 = (blockIdx.x & 127) * 16;
  const bf16_t* Qp  = Qb + (size_t)b * S_LEN * D_DIM;
  const bf16_t* Kp  = Kb + (size_t)b * S_LEN * D_DIM;
  const bf16_t* Vtp = Vt + (size_t)b * D_DIM * S_LEN;   // [D][S]

  if (threadIdx.x < 32) {
    tdm_load_2d_bf16(0u, Qp + (size_t)q0 * D_DIM, 16u, (unsigned)D_DIM,
                     (unsigned)D_DIM);
    __builtin_amdgcn_s_wait_tensorcnt(0);
  }
  if (threadIdx.x < 16) lsum[threadIdx.x] = 0.f;
  __syncthreads();

  v8f acc[8] = {};
  const float scale = 0.03125f;                 // 1/sqrt(1024)
  const int mOff = (lane >> 4) ? 8 : 0;
  const int qmax = q0 + 15;

  for (int kb = 0; kb <= qmax; kb += 128) {
    const int k0w = kb + wave * 16;
    if (kb + 128 <= qmax)
      __builtin_prefetch(Kp + (size_t)(k0w + 128) * D_DIM, 0, 1);
    // ---- scores for this wave's 16 keys ----
    v8f s = {};
#pragma unroll
    for (int dk = 0; dk < 32; ++dk) {
      v16bf aQ = load_a_bf16(Qlds + dk * 32, D_DIM);
      v16bf bk = load_bT_bf16(Kp + (size_t)k0w * D_DIM + dk * 32, D_DIM);
      s = wmma_bf16(aQ, bk, s);
    }
    const int key = k0w + (lane & 15);
#pragma unroll
    for (int r = 0; r < 8; ++r) {
      const int m = r + mOff;
      float e = (key <= q0 + m) ? __expf(s[r] * scale) : 0.f;
      Plds[m * 128 + wave * 16 + (lane & 15)] = f2bf(e);
      float rs = e;
      rs += __shfl_xor(rs, 1);
      rs += __shfl_xor(rs, 2);
      rs += __shfl_xor(rs, 4);
      rs += __shfl_xor(rs, 8);
      if ((lane & 15) == 0) atomicAdd(&lsum[m], rs);
    }
    __syncthreads();
    // ---- P @ V into this wave's 16x128 d-slab (B-fragments from V^T) ----
#pragma unroll
    for (int ks = 0; ks < 4; ++ks) {
      v16bf aP = load_a_bf16(Plds + ks * 32, 128);
#pragma unroll
      for (int sub = 0; sub < 8; ++sub) {
        const int n0 = wave * 128 + sub * 16;
        v16bf bv = load_bT_bf16(Vtp + (size_t)n0 * S_LEN + kb + ks * 32, S_LEN);
        acc[sub] = wmma_bf16(aP, bv, acc[sub]);
      }
    }
    __syncthreads();
  }

#pragma unroll
  for (int r = 0; r < 8; ++r) {
    const int m = r + mOff;
    const float inv = 1.0f / lsum[m];
#pragma unroll
    for (int sub = 0; sub < 8; ++sub) {
      const int n = wave * 128 + sub * 16 + (lane & 15);
      out[((size_t)b * S_LEN + q0 + m) * D_DIM + n] = acc[sub][r] * inv;
    }
  }
}

extern "C" void kernel_launch(void* const* d_in, const int* in_sizes, int n_in,
                              void* d_out, int out_size, void* d_ws, size_t ws_size,
                              hipStream_t stream) {
  (void)in_sizes; (void)n_in; (void)out_size; (void)ws_size;
  const float* x  = (const float*)d_in[0];
  const float* Wq = (const float*)d_in[1];
  const float* bq = (const float*)d_in[2];
  const float* Wk = (const float*)d_in[3];
  const float* bk = (const float*)d_in[4];
  const float* Wv = (const float*)d_in[5];
  const float* bv = (const float*)d_in[6];
  float* out = (float*)d_out;

  const size_t BSD = (size_t)BATCH * S_LEN * D_DIM;   // 8M
  const size_t DD  = (size_t)D_DIM * D_DIM;           // 1M
  bf16_t* xb  = (bf16_t*)d_ws;
  bf16_t* Wqt = xb + BSD;
  bf16_t* Wkt = Wqt + DD;
  bf16_t* Wvt = Wkt + DD;
  bf16_t* Qb  = Wvt + DD;
  bf16_t* Kb  = Qb + BSD;
  bf16_t* Vt  = Kb + BSD;                             // [B][D][S]

  xconv_kernel<<<(int)(BSD / 2048), 256, 0, stream>>>(x, xb);
  wconv_kernel<<<dim3(256, 3), 256, 0, stream>>>(Wq, Wk, Wv, Wqt, Wkt, Wvt);
  qkv_kernel<<<512, 256, 0, stream>>>(xb, Wqt, Wkt, Wvt, bq, bk, bv, Qb, Kb, Vt);
  attn_kernel<<<512, 256, 0, stream>>>(Qb, Kb, Vt, out);
}
